// Decoder_28338194219011
// MI455X (gfx1250) — matrix operand order repaired and checked by reading, not compiled, not run on hardware
//
#include <hip/hip_runtime.h>

// ---------------------------------------------------------------------------
// Problem constants (match the reference)
// ---------------------------------------------------------------------------
static const int B = 32, T = 128, S = 128, H = 1024;
static const int G4H = 4 * H;          // 4096 gate width
static const long long BT = (long long)B * T;

typedef __attribute__((ext_vector_type(16))) __bf16         v16bf;
typedef __attribute__((ext_vector_type(8)))  float          v8f;
typedef __attribute__((ext_vector_type(8)))  unsigned short us8;

union Frag { v16bf v; us8 h[2]; };

__device__ __forceinline__ unsigned short f2bf(float f) {
    unsigned u = __float_as_uint(f);
    u += 0x7FFFu + ((u >> 16) & 1u);          // round-to-nearest-even
    return (unsigned short)(u >> 16);
}
__device__ __forceinline__ float sigf(float x) { return 1.0f / (1.0f + __expf(-x)); }

// A fragment: 16x32 bf16 tile, A row-major, tile origin passed in.
// ISA layout: lane L holds row m=L&15; half=L>>4 selects K {h*8..h*8+7, 16+h*8..16+h*8+7}
__device__ __forceinline__ v16bf load_frag_a(const unsigned short* A, int lda) {
    const int lane = threadIdx.x & 31;
    const unsigned short* p = A + (long long)(lane & 15) * lda + (lane >> 4) * 8;
    Frag f;
    f.h[0] = *(const us8*)(p);
    f.h[1] = *(const us8*)(p + 16);
    return f.v;
}
// B fragment: 32x16 bf16 tile where B[k][n] = Bt[n][k] (Bt row-major, ld=ldb).
// ISA layout: lane L holds column n=L&15; half=L>>4 selects K {h*8..h*8+7, 16+h*8..16+h*8+7}
__device__ __forceinline__ v16bf load_frag_b(const unsigned short* Bt, int ldb) {
    const int lane = threadIdx.x & 31;
    const unsigned short* p = Bt + (long long)(lane & 15) * ldb + (lane >> 4) * 8;
    Frag f;
    f.h[0] = *(const us8*)(p);
    f.h[1] = *(const us8*)(p + 16);
    return f.v;
}

// ---------------------------------------------------------------------------
// Generic tiled GEMM:  C[M,N] = A[M,K](bf16) * Bt[N,K](bf16)^T
// One wave computes a 16x64 strip (4 accumulators, A-frag reused 4x per K step).
// epi: 0 = store f32, 1 = store bf16, 2 = store f32 tanh(acc + D + bias)
// blockIdx.z = batch, with element strides sA/sB/sC.
// ---------------------------------------------------------------------------
__global__ __launch_bounds__(256)
void gemm_bt_kernel(const unsigned short* __restrict__ A,  long long sA, int lda,
                    const unsigned short* __restrict__ Bt, long long sB, int ldb,
                    void* __restrict__ Cv, long long sC, int ldc,
                    const float* __restrict__ Dadd, int ldd,
                    const float* __restrict__ bias,
                    int M, int N, int K, int epi)
{
    const int nT64  = N >> 6;
    const int tiles = (M >> 4) * nT64;
    const int wid   = blockIdx.x * 8 + ((int)threadIdx.x >> 5);
    if (wid >= tiles) return;                       // whole-wave exit: EXEC all-1 for WMMA
    const int batch = blockIdx.z;
    A  += (long long)batch * sA;
    Bt += (long long)batch * sB;

    const int mT = wid / nT64;
    const int nT = wid % nT64;
    const unsigned short* Ar = A  + (long long)(mT << 4) * lda;
    const unsigned short* Bb = Bt + (long long)(nT << 6) * ldb;

    v8f acc0 = {}, acc1 = {}, acc2 = {}, acc3 = {};
    for (int k = 0; k < K; k += 32) {
        v16bf a  = load_frag_a(Ar + k, lda);
        v16bf b0 = load_frag_b(Bb + k, ldb);
        v16bf b1 = load_frag_b(Bb + (long long)16 * ldb + k, ldb);
        v16bf b2 = load_frag_b(Bb + (long long)32 * ldb + k, ldb);
        v16bf b3 = load_frag_b(Bb + (long long)48 * ldb + k, ldb);
        acc0 = __builtin_amdgcn_wmma_f32_16x16x32_bf16(false, a, false, b0, (short)0, acc0, false, false);
        acc1 = __builtin_amdgcn_wmma_f32_16x16x32_bf16(false, a, false, b1, (short)0, acc1, false, false);
        acc2 = __builtin_amdgcn_wmma_f32_16x16x32_bf16(false, a, false, b2, (short)0, acc2, false, false);
        acc3 = __builtin_amdgcn_wmma_f32_16x16x32_bf16(false, a, false, b3, (short)0, acc3, false, false);
    }

    const int lane = threadIdx.x & 31;
    const int cn   = lane & 15;
    const int rb   = (lane >> 4) * 8;
#pragma unroll
    for (int r = 0; r < 8; ++r) {
        const int row = (mT << 4) + rb + r;
        const int c0  = (nT << 6) + cn;
        const float v0 = acc0[r], v1 = acc1[r], v2 = acc2[r], v3 = acc3[r];
        if (epi == 0) {
            float* C = (float*)Cv + (long long)batch * sC + (long long)row * ldc;
            C[c0] = v0; C[c0 + 16] = v1; C[c0 + 32] = v2; C[c0 + 48] = v3;
        } else if (epi == 1) {
            unsigned short* C = (unsigned short*)Cv + (long long)batch * sC + (long long)row * ldc;
            C[c0] = f2bf(v0); C[c0 + 16] = f2bf(v1); C[c0 + 32] = f2bf(v2); C[c0 + 48] = f2bf(v3);
        } else {
            float* C = (float*)Cv + (long long)row * ldc;
            const float* Dr = Dadd + (long long)row * ldd;
            C[c0]      = tanhf(v0 + Dr[c0]      + bias[c0]);
            C[c0 + 16] = tanhf(v1 + Dr[c0 + 16] + bias[c0 + 16]);
            C[c0 + 32] = tanhf(v2 + Dr[c0 + 32] + bias[c0 + 32]);
            C[c0 + 48] = tanhf(v3 + Dr[c0 + 48] + bias[c0 + 48]);
        }
    }
}

// ---------------------------------------------------------------------------
// Prep / elementwise kernels
// ---------------------------------------------------------------------------
__global__ void cast_bf_kernel(const float* __restrict__ src, unsigned short* __restrict__ dst,
                               long long n) {
    long long i = (long long)blockIdx.x * blockDim.x + threadIdx.x;
    if (i < n) dst[i] = f2bf(src[i]);
}

__global__ void embed_cast_kernel(const float* __restrict__ emb, const int* __restrict__ tgt,
                                  unsigned short* __restrict__ xbf) {
    long long i = (long long)blockIdx.x * blockDim.x + threadIdx.x;
    if (i >= BT * H) return;
    const long long bt = i / H;
    const int       hh = (int)(i % H);
    xbf[i] = f2bf(emb[(long long)tgt[bt] * H + hh]);
}

__global__ void enc_prep_kernel(const float* __restrict__ enc,
                                unsigned short* __restrict__ enc_bf,
                                unsigned short* __restrict__ encT_bf) {
    long long i = (long long)blockIdx.x * blockDim.x + threadIdx.x;
    if (i >= (long long)B * S * H) return;
    const int b  = (int)(i / ((long long)S * H));
    const int r  = (int)(i % ((long long)S * H));
    const int s  = r / H;
    const int hh = r % H;
    const unsigned short v = f2bf(enc[i]);
    enc_bf[i] = v;
    encT_bf[((long long)b * H + hh) * S + s] = v;
}

__global__ void bias_sum_kernel(const float* __restrict__ a, const float* __restrict__ b,
                                float* __restrict__ o, int n) {
    int i = blockIdx.x * blockDim.x + threadIdx.x;
    if (i < n) o[i] = a[i] + b[i];
}

// One wave per row (S=128 -> 4 values/lane). Masked softmax -> bf16 attn.
__global__ __launch_bounds__(256)
void softmax_kernel(const float* __restrict__ scores, const int* __restrict__ src_len,
                    unsigned short* __restrict__ attn_bf) {
    const int row = blockIdx.x * 8 + ((int)threadIdx.x >> 5);
    if (row >= (int)BT) return;
    const int L    = src_len[row / T];
    const int lane = threadIdx.x & 31;
    const float* sr = scores + (long long)row * S;
    float v[4]; float mx = -3.0e38f;
#pragma unroll
    for (int j = 0; j < 4; ++j) {
        const int s = lane + j * 32;
        v[j] = (s < L) ? sr[s] : -1.0e9f;
        mx = fmaxf(mx, v[j]);
    }
#pragma unroll
    for (int off = 16; off > 0; off >>= 1) mx = fmaxf(mx, __shfl_xor(mx, off, 32));
    float sum = 0.0f;
#pragma unroll
    for (int j = 0; j < 4; ++j) { v[j] = __expf(v[j] - mx); sum += v[j]; }
#pragma unroll
    for (int off = 16; off > 0; off >>= 1) sum += __shfl_xor(sum, off, 32);
    const float inv = 1.0f / sum;
    unsigned short* ar = attn_bf + (long long)row * S;
#pragma unroll
    for (int j = 0; j < 4; ++j) ar[lane + j * 32] = f2bf(v[j] * inv);
}

__global__ void init_state_kernel(const float* __restrict__ h0, const float* __restrict__ c0,
                                  float* __restrict__ h, float* __restrict__ c,
                                  unsigned short* __restrict__ hbf) {
    int i = blockIdx.x * blockDim.x + threadIdx.x;
    if (i >= B * H) return;
    h[i] = h0[i]; c[i] = c0[i]; hbf[i] = f2bf(h0[i]);
}

// gates = gates_h + gates_in[:, t, :] + (bih+bhh);  PyTorch i,f,g,o order.
__global__ void lstm_cell_kernel(const float* __restrict__ gh, const float* __restrict__ gates_in,
                                 const float* __restrict__ bvec,
                                 float* __restrict__ c, float* __restrict__ h,
                                 unsigned short* __restrict__ hbf, int t) {
    int i = blockIdx.x * blockDim.x + threadIdx.x;
    if (i >= B * H) return;
    const int b = i / H, n = i % H;
    const float* g  = gh + (long long)b * G4H;
    const float* gi = gates_in + ((long long)b * T + t) * G4H;
    const float ig = g[n]         + gi[n]         + bvec[n];
    const float fg = g[H + n]     + gi[H + n]     + bvec[H + n];
    const float gg = g[2 * H + n] + gi[2 * H + n] + bvec[2 * H + n];
    const float og = g[3 * H + n] + gi[3 * H + n] + bvec[3 * H + n];
    const float cn = sigf(fg) * c[i] + sigf(ig) * tanhf(gg);
    const float hn = sigf(og) * tanhf(cn);
    c[i] = cn; h[i] = hn; hbf[i] = f2bf(hn);
}

__global__ void final_copy_kernel(const float* __restrict__ h, const float* __restrict__ c,
                                  float* __restrict__ out) {
    int i = blockIdx.x * blockDim.x + threadIdx.x;
    if (i >= B * H) return;
    out[BT * H + i]         = h[i];
    out[BT * H + B * H + i] = c[i];
}

// ---------------------------------------------------------------------------
// Host orchestration
// ---------------------------------------------------------------------------
static inline int cdiv(long long a, long long b) { return (int)((a + b - 1) / b); }

extern "C" void kernel_launch(void* const* d_in, const int* in_sizes, int n_in,
                              void* d_out, int out_size, void* d_ws, size_t ws_size,
                              hipStream_t stream)
{
    const float* emb = (const float*)d_in[0];
    const float* Wih = (const float*)d_in[1];
    const float* Whh = (const float*)d_in[2];
    const float* bih = (const float*)d_in[3];
    const float* bhh = (const float*)d_in[4];
    const float* Wfc = (const float*)d_in[5];
    const float* bfc = (const float*)d_in[6];
    const int*   tgt = (const int*)d_in[7];
    const float* h0  = (const float*)d_in[8];
    const float* c0  = (const float*)d_in[9];
    const float* enc = (const float*)d_in[10];
    const int* srcl  = (const int*)d_in[11];
    float* out = (float*)d_out;

    // workspace carve-up (256B aligned chunks)
    char* p = (char*)d_ws;
    auto alloc = [&](size_t bytes) { char* r = p; p += (bytes + 255) & ~(size_t)255; return r; };
    unsigned short* x_bf    = (unsigned short*)alloc(BT * H * 2);        // 8 MB
    unsigned short* enc_bf  = (unsigned short*)alloc((size_t)B * S * H * 2);
    unsigned short* encT_bf = (unsigned short*)alloc((size_t)B * H * S * 2);
    unsigned short* Wih_bf  = (unsigned short*)alloc((size_t)G4H * H * 2);
    unsigned short* Whh_bf  = (unsigned short*)alloc((size_t)G4H * H * 2);
    unsigned short* Wfc_bf  = (unsigned short*)alloc((size_t)H * 2 * H * 2);
    float*          bvec    = (float*)alloc(G4H * 4);
    float*          scores  = (float*)alloc(BT * S * 4);                 // 2 MB
    unsigned short* attn_bf = (unsigned short*)alloc(BT * S * 2);
    unsigned short* ctx_bf  = (unsigned short*)alloc(BT * H * 2);
    float*          fcc     = (float*)alloc(BT * H * 4);                 // 16 MB
    float*          gatesin = (float*)alloc(BT * G4H * 4);               // 64 MB
    float*          gatesh  = (float*)alloc((size_t)B * G4H * 4);
    float*          hbuf    = (float*)alloc((size_t)B * H * 4);
    float*          cbuf    = (float*)alloc((size_t)B * H * 4);
    unsigned short* h_bf    = (unsigned short*)alloc((size_t)B * H * 2);

    // --- prep: casts / gather / transpose -----------------------------------
    embed_cast_kernel<<<cdiv(BT * H, 256), 256, 0, stream>>>(emb, tgt, x_bf);
    enc_prep_kernel<<<cdiv((long long)B * S * H, 256), 256, 0, stream>>>(enc, enc_bf, encT_bf);
    cast_bf_kernel<<<cdiv((long long)G4H * H, 256), 256, 0, stream>>>(Wih, Wih_bf, (long long)G4H * H);
    cast_bf_kernel<<<cdiv((long long)G4H * H, 256), 256, 0, stream>>>(Whh, Whh_bf, (long long)G4H * H);
    cast_bf_kernel<<<cdiv((long long)H * 2 * H, 256), 256, 0, stream>>>(Wfc, Wfc_bf, (long long)H * 2 * H);
    bias_sum_kernel<<<cdiv(G4H, 256), 256, 0, stream>>>(bih, bhh, bvec, G4H);

    // --- attention: scores = x @ enc^T (batched over B) ---------------------
    {   // per batch: M=T, N=S, K=H -> 8*2 = 16 tiles -> grid.x = 2
        dim3 g(cdiv((T / 16) * (S / 64), 8), 1, B);
        gemm_bt_kernel<<<g, 256, 0, stream>>>(x_bf, (long long)T * H, H,
                                              enc_bf, (long long)S * H, H,
                                              scores, (long long)T * S, S,
                                              nullptr, 0, nullptr, T, S, H, 0);
    }
    softmax_kernel<<<cdiv(BT, 8), 256, 0, stream>>>(scores, srcl, attn_bf);
    // --- ctx = attn @ enc  (Bt = encT), bf16 output -------------------------
    {   // per batch: M=T, N=H, K=S -> 8*16 = 128 tiles -> grid.x = 16
        dim3 g(cdiv((T / 16) * (H / 64), 8), 1, B);
        gemm_bt_kernel<<<g, 256, 0, stream>>>(attn_bf, (long long)T * S, S,
                                              encT_bf, (long long)H * S, S,
                                              ctx_bf, (long long)T * H, H,
                                              nullptr, 0, nullptr, T, H, S, 1);
    }
    // --- hoisted GEMMs: gates_in = x @ Wih^T ; fcc = ctx @ Wfc[:, :H]^T -----
    {   dim3 g(cdiv(((int)BT / 16) * (G4H / 64), 8), 1, 1);
        gemm_bt_kernel<<<g, 256, 0, stream>>>(x_bf, 0, H, Wih_bf, 0, H,
                                              gatesin, 0, G4H,
                                              nullptr, 0, nullptr, (int)BT, G4H, H, 0);
    }
    {   dim3 g(cdiv(((int)BT / 16) * (H / 64), 8), 1, 1);
        gemm_bt_kernel<<<g, 256, 0, stream>>>(ctx_bf, 0, H, Wfc_bf, 0, 2 * H,
                                              fcc, 0, H,
                                              nullptr, 0, nullptr, (int)BT, H, H, 0);
    }

    // --- sequential LSTM recurrence ----------------------------------------
    init_state_kernel<<<cdiv(B * H, 256), 256, 0, stream>>>(h0, c0, hbuf, cbuf, h_bf);
    for (int t = 0; t < T; ++t) {
        // gates_h = h @ Whh^T : M=32, N=4096, K=1024 -> 2*64 = 128 tiles
        dim3 g1(cdiv((B / 16) * (G4H / 64), 8), 1, 1);
        gemm_bt_kernel<<<g1, 256, 0, stream>>>(h_bf, 0, H, Whh_bf, 0, H,
                                               gatesh, 0, G4H,
                                               nullptr, 0, nullptr, B, G4H, H, 0);
        lstm_cell_kernel<<<cdiv(B * H, 256), 256, 0, stream>>>(gatesh, gatesin, bvec,
                                                               cbuf, hbuf, h_bf, t);
        // out[:, t, :] = tanh(fcc[:, t, :] + h @ Wfc[:, H:]^T + bfc), fused epilogue
        dim3 g2(cdiv((B / 16) * (H / 64), 8), 1, 1);
        gemm_bt_kernel<<<g2, 256, 0, stream>>>(h_bf, 0, H, Wfc_bf + H, 0, 2 * H,
                                               out + (long long)t * H, 0, T * H,
                                               fcc + (long long)t * H, T * H, bfc,
                                               B, H, H, 2);
    }
    final_copy_kernel<<<cdiv(B * H, 256), 256, 0, stream>>>(hbuf, cbuf, out);
}
